// ScannedRNN_61168924229650
// MI455X (gfx1250) — matrix operand order repaired and checked by reading, not compiled, not run on hardware
//
#include <hip/hip_runtime.h>
#include <cstdint>
#include <cstddef>

// ---------------------------------------------------------------------------
// GRU scan for MI455X (gfx1250): bf16 WMMA (f32 accum) for all GEMMs.
//   T=512, B=128, D=512, H=512, 3H=1536
// Phase 1: pack Wi and [Wh_rz | Wh_n] into B-fragment-native bf16 layout.
// Phase 2: x_proj = ins @ Wi + bi  (WMMA; 3 interleaved accumulator chains,
//          fragments double-buffered one K-step ahead).
// Phase 3: 512 sequential gru_step launches; each wave owns an (r,z,n)
//          column-tile triple; same pipelining; gates fused in registers,
//          carry lives in ys[t-1].
// ---------------------------------------------------------------------------

typedef __attribute__((ext_vector_type(16))) __bf16 v16bf;
typedef __attribute__((ext_vector_type(8)))  float  v8f;

#define T_STEPS 512
#define Bm      128
#define Dk      512
#define Hh      512
#define NCOLS   1536   // 3H
#define NT      96     // 1536/16 column tiles
#define KK      16     // 512/32 K-steps

union BFrag {
  uint4   q[2];
  v16bf   v;
};

__device__ __forceinline__ unsigned short f2bf(float f) {
  union { float f; uint32_t u; } c; c.f = f;
  uint32_t r = c.u + 0x7FFFu + ((c.u >> 16) & 1u);  // round-to-nearest-even
  return (unsigned short)(r >> 16);
}

__device__ __forceinline__ float sigmoidf_(float x) {
  return 1.0f / (1.0f + __expf(-x));
}

__device__ __forceinline__ void loadB(const unsigned short* __restrict__ W,
                                      uint32_t kk, uint32_t nt, uint32_t lane,
                                      BFrag& f) {
  const uint4* p = (const uint4*)(W + (((size_t)kk * NT + nt) * 32 + lane) * 16);
  f.q[0] = p[0];
  f.q[1] = p[1];
}

__device__ __forceinline__ void loadA(const unsigned short* hA, uint32_t stride,
                                      uint32_t kk, uint32_t arow, uint32_t koff,
                                      BFrag& f) {
  const uint4* p = (const uint4*)&hA[arow * stride + kk * 32 + koff];
  f.q[0] = p[0];   // K = base .. base+7
  f.q[1] = p[1];   // K = base+8 .. base+15
}

// ---------------------------------------------------------------------------
// Pack f32 weights into bf16 B-fragment layout:
//   dst[((kk*NT + nt)*32 + lane)*16 + e] = W[kk*32 + (lane>>4)*16 + e][nt*16 + (lane&15)]
// ---------------------------------------------------------------------------
__global__ void pack_weights(const float* __restrict__ Wi,
                             const float* __restrict__ Wh_rz,
                             const float* __restrict__ Wh_n,
                             unsigned short* __restrict__ WiB,
                             unsigned short* __restrict__ WhB) {
  uint32_t id = blockIdx.x * blockDim.x + threadIdx.x;
  const uint32_t per = Dk * NCOLS;                 // 786432 elements per weight
  uint32_t w    = id / per;
  uint32_t rem  = id - w * per;
  uint32_t e    = rem & 15u;
  uint32_t lane = (rem >> 4) & 31u;
  uint32_t tile = rem >> 9;
  uint32_t nt   = tile % NT;
  uint32_t kk   = tile / NT;
  uint32_t k    = kk * 32 + (lane >> 4) * 16 + e;
  uint32_t n    = nt * 16 + (lane & 15u);
  float v;
  if (w == 0) {
    v = Wi[(size_t)k * NCOLS + n];
  } else {
    v = (n < 2u * Hh) ? Wh_rz[(size_t)k * (2u * Hh) + n]
                      : Wh_n[(size_t)k * Hh + (n - 2u * Hh)];
  }
  (w == 0 ? WiB : WhB)[rem] = f2bf(v);
}

// ---------------------------------------------------------------------------
// x_proj = ins @ Wi + bi.  Block = 16-row slab of [T*B, D]; 8 waves cover
// all 96 column tiles (4 triples each); fragments double-buffered.
// ---------------------------------------------------------------------------
__global__ void __launch_bounds__(256)
xproj_gemm(const float* __restrict__ ins,
           const unsigned short* __restrict__ WiB,
           const float* __restrict__ bi,
           float* __restrict__ xp) {
  __shared__ __align__(16) unsigned short hA[16 * Dk];   // 16 KB
  const uint32_t tid = threadIdx.x;
  const uint32_t m0  = blockIdx.x * 16;

  for (uint32_t idx = tid; idx < 16u * Dk; idx += 256u) {
    uint32_t row = idx >> 9, col = idx & 511u;
    hA[idx] = f2bf(ins[(size_t)(m0 + row) * Dk + col]);
  }
  __syncthreads();

  const uint32_t wave = tid >> 5, lane = tid & 31u;
  const uint32_t arow = lane & 15u;
  const uint32_t koff = (lane >> 4) * 16;   // ushort offset of first K chunk

  for (uint32_t i = 0; i < 4; ++i) {
    const uint32_t nt0 = wave * 12 + 3 * i;   // triple (nt0, nt0+1, nt0+2)
    v8f acc0 = {}, acc1 = {}, acc2 = {};
    BFrag a[2], b0[2], b1[2], b2[2];
    loadA(hA, Dk, 0, arow, koff, a[0]);
    loadB(WiB, 0, nt0,     lane, b0[0]);
    loadB(WiB, 0, nt0 + 1, lane, b1[0]);
    loadB(WiB, 0, nt0 + 2, lane, b2[0]);
    #pragma unroll
    for (uint32_t kk = 0; kk < KK; ++kk) {
      const uint32_t cur = kk & 1u, nxt = cur ^ 1u;
      if (kk + 1 < KK) {
        loadA(hA, Dk, kk + 1, arow, koff, a[nxt]);
        loadB(WiB, kk + 1, nt0,     lane, b0[nxt]);
        loadB(WiB, kk + 1, nt0 + 1, lane, b1[nxt]);
        loadB(WiB, kk + 1, nt0 + 2, lane, b2[nxt]);
      }
      acc0 = __builtin_amdgcn_wmma_f32_16x16x32_bf16(
          false, a[cur].v, false, b0[cur].v, (short)0, acc0, false, false);
      acc1 = __builtin_amdgcn_wmma_f32_16x16x32_bf16(
          false, a[cur].v, false, b1[cur].v, (short)0, acc1, false, false);
      acc2 = __builtin_amdgcn_wmma_f32_16x16x32_bf16(
          false, a[cur].v, false, b2[cur].v, (short)0, acc2, false, false);
    }
    #pragma unroll
    for (uint32_t j = 0; j < 3; ++j) {
      const v8f acc = (j == 0) ? acc0 : (j == 1) ? acc1 : acc2;
      const uint32_t col = (nt0 + j) * 16 + (lane & 15u);
      const float bias = bi[col];
      #pragma unroll
      for (uint32_t r = 0; r < 8; ++r) {
        uint32_t row = r + 8 * (lane >> 4);
        xp[(size_t)(m0 + row) * NCOLS + col] = acc[r] + bias;
      }
    }
  }
}

// ---------------------------------------------------------------------------
// One GRU step. grid = (8 row tiles, 4 col groups), block = 256 (8 waves).
// Wave owns r/z/n tile triple (nt, nt+32, nt+64); fragments double-buffered;
// gate math fused in registers; result -> ys[t].
// ---------------------------------------------------------------------------
__global__ void __launch_bounds__(256)
gru_step(const float* __restrict__ xp_t,              // [B, 3H] slice
         const unsigned short* __restrict__ WhB,
         const unsigned char* __restrict__ resets_t,  // [B]
         const float* __restrict__ h_prev,            // [B, H]
         const float* __restrict__ bh_n,              // [H]
         float* __restrict__ ys_t) {                  // [B, H]
  __shared__ __align__(16) unsigned short hA[16 * Hh];  // 16 KB
  const uint32_t tid = threadIdx.x;
  const uint32_t m0  = blockIdx.x * 16;

  // Masked carry -> bf16 LDS tile (A matrix slab).
  for (uint32_t idx = tid; idx < 16u * Hh; idx += 256u) {
    uint32_t row = idx >> 9, col = idx & 511u;
    uint32_t b   = m0 + row;
    float v = resets_t[b] ? 0.0f : h_prev[(size_t)b * Hh + col];
    hA[idx] = f2bf(v);
  }
  __syncthreads();

  const uint32_t wave = tid >> 5, lane = tid & 31u;
  const uint32_t nt   = blockIdx.y * 8 + wave;        // 0..31 (r columns)
  const uint32_t arow = lane & 15u;
  const uint32_t koff = (lane >> 4) * 16;

  v8f accR = {}, accZ = {}, accN = {};
  BFrag a[2], br[2], bz[2], bn[2];
  loadA(hA, Hh, 0, arow, koff, a[0]);
  loadB(WhB, 0, nt,      lane, br[0]);
  loadB(WhB, 0, nt + 32, lane, bz[0]);
  loadB(WhB, 0, nt + 64, lane, bn[0]);
  #pragma unroll
  for (uint32_t kk = 0; kk < KK; ++kk) {
    const uint32_t cur = kk & 1u, nxt = cur ^ 1u;
    if (kk + 1 < KK) {
      loadA(hA, Hh, kk + 1, arow, koff, a[nxt]);
      loadB(WhB, kk + 1, nt,      lane, br[nxt]);
      loadB(WhB, kk + 1, nt + 32, lane, bz[nxt]);
      loadB(WhB, kk + 1, nt + 64, lane, bn[nxt]);
    }
    accR = __builtin_amdgcn_wmma_f32_16x16x32_bf16(
        false, a[cur].v, false, br[cur].v, (short)0, accR, false, false);
    accZ = __builtin_amdgcn_wmma_f32_16x16x32_bf16(
        false, a[cur].v, false, bz[cur].v, (short)0, accZ, false, false);
    accN = __builtin_amdgcn_wmma_f32_16x16x32_bf16(
        false, a[cur].v, false, bn[cur].v, (short)0, accN, false, false);
  }

  // Fused gate math; C/D layout: element r, lane L -> row r+8*(L>>4), col L&15.
  const uint32_t c    = nt * 16 + (lane & 15u);
  const float    bhn  = bh_n[c];
  #pragma unroll
  for (uint32_t r = 0; r < 8; ++r) {
    uint32_t row = r + 8 * (lane >> 4);
    uint32_t b   = m0 + row;
    const float* xrow = xp_t + (size_t)b * NCOLS;
    float rr = sigmoidf_(xrow[c]            + accR[r]);
    float zz = sigmoidf_(xrow[Hh + c]       + accZ[r]);
    float nn = tanhf    (xrow[2u * Hh + c]  + rr * (accN[r] + bhn));
    float hm = resets_t[b] ? 0.0f : h_prev[(size_t)b * Hh + c];
    ys_t[(size_t)b * Hh + c] = (1.0f - zz) * nn + zz * hm;
  }
}

// ---------------------------------------------------------------------------
// Host side. Inputs: ins, resets, h0, Wi, bi, Wh_rz, Wh_n, bh_n.
// Output: hT [B,H] then ys [T,B,H], fp32, concatenated flat.
// Workspace: WiB (1.5MB bf16) | WhB (1.5MB bf16) | x_proj (403MB f32).
// ---------------------------------------------------------------------------
extern "C" void kernel_launch(void* const* d_in, const int* in_sizes, int n_in,
                              void* d_out, int out_size, void* d_ws, size_t ws_size,
                              hipStream_t stream) {
  const float*         ins    = (const float*)d_in[0];
  const unsigned char* resets = (const unsigned char*)d_in[1];
  const float*         h0     = (const float*)d_in[2];
  const float*         Wi     = (const float*)d_in[3];
  const float*         bi     = (const float*)d_in[4];
  const float*         Wh_rz  = (const float*)d_in[5];
  const float*         Wh_n   = (const float*)d_in[6];
  const float*         bh_n   = (const float*)d_in[7];

  float* out = (float*)d_out;
  float* hT  = out;                                 // [B, H]
  float* ys  = out + (size_t)Bm * Hh;               // [T, B, H]

  char* ws = (char*)d_ws;
  unsigned short* WiB = (unsigned short*)ws;
  unsigned short* WhB = (unsigned short*)(ws + (size_t)Dk * NCOLS * 2);
  float*          xp  = (float*)(ws + (size_t)Dk * NCOLS * 4);

  // Phase 1: pack both weight matrices (2 * 786432 elements).
  pack_weights<<<(2u * Dk * NCOLS) / 256u, 256, 0, stream>>>(
      Wi, Wh_rz, Wh_n, WiB, WhB);

  // Phase 2: hoisted input GEMM.
  xproj_gemm<<<(T_STEPS * Bm) / 16, 256, 0, stream>>>(ins, WiB, bi, xp);

  // Phase 3: sequential scan; carry lives in ys[t-1].
  for (int t = 0; t < T_STEPS; ++t) {
    const float* hp = (t == 0) ? h0 : (ys + (size_t)(t - 1) * Bm * Hh);
    gru_step<<<dim3(Bm / 16, 4), 256, 0, stream>>>(
        xp + (size_t)t * Bm * NCOLS, WhB, resets + (size_t)t * Bm, hp, bh_n,
        ys + (size_t)t * Bm * Hh);
  }

  // hT = final carry = ys[T-1].
  hipMemcpyAsync(hT, ys + (size_t)(T_STEPS - 1) * Bm * Hh,
                 (size_t)Bm * Hh * sizeof(float),
                 hipMemcpyDeviceToDevice, stream);
}
